// PAMNet_17557826306684
// MI455X (gfx1250) — matrix operand order repaired and checked by reading, not compiled, not run on hardware
//
#include <hip/hip_runtime.h>
#include <math.h>

// ---------------------------------------------------------------------------
// Types for CDNA5 WMMA (wave32). v16h A/B fragments, v8f accumulator.
// ---------------------------------------------------------------------------
typedef _Float16 f16;
typedef f16  half16 __attribute__((ext_vector_type(16)));
typedef float v8f   __attribute__((ext_vector_type(8)));

// Pointer types matching __builtin_amdgcn_global_load_async_to_lds_b128:
//   (int4 addrspace(1)*, int4 addrspace(3)*, imm offset, imm cpol)
typedef int i32x4 __attribute__((vector_size(16)));
typedef __attribute__((address_space(1))) i32x4* gbl_i32x4p;
typedef __attribute__((address_space(3))) i32x4* lds_i32x4p;
typedef __attribute__((address_space(3))) f16*   as3f16p;

#define TKP 40   // LDS row stride in f16 elements (32 + 8 pad; 80B rows, 16B aligned)

// ---------------------------------------------------------------------------
// Activation codes: 0 none, 1 relu, 2 gelu(tanh), 3 softplus, 4 silu, 5 elu
// ---------------------------------------------------------------------------
__device__ __forceinline__ float apply_act(float v, int act) {
  switch (act) {
    case 1: return fmaxf(v, 0.f);
    case 2: { float t = 0.7978845608028654f * (v + 0.044715f * v * v * v);
              return 0.5f * v * (1.f + tanhf(t)); }
    case 3: return (v > 20.f) ? v : log1pf(expf(v));
    case 4: return v / (1.f + expf(-v));
    case 5: return (v > 0.f) ? v : (expf(v) - 1.f);
  }
  return v;
}

// Async global -> LDS 16-byte copy (CDNA5 async data path, ASYNCcnt).
__device__ __forceinline__ void async_copy16(const f16* gp, f16* sp) {
#if __has_builtin(__builtin_amdgcn_global_load_async_to_lds_b128)
  __builtin_amdgcn_global_load_async_to_lds_b128(
      (gbl_i32x4p)(f16*)gp, (lds_i32x4p)sp, 0, 0);
#else
  unsigned off = (unsigned)(size_t)(as3f16p)sp;
  asm volatile("global_load_async_to_lds_b128 %0, %1, off"
               :: "v"(off), "v"(gp) : "memory");
#endif
}

__device__ __forceinline__ void wait_async0() {
#if __has_builtin(__builtin_amdgcn_s_wait_asynccnt)
  __builtin_amdgcn_s_wait_asynccnt(0);
#else
  asm volatile("s_wait_asynccnt 0" ::: "memory");
#endif
}

// Fast path: full 128-row tile, full 32-wide K slice -> pure async copies.
__device__ __forceinline__ void g2lds_tile_async(const f16* __restrict__ G, long ld,
                                                 long row0, int k0,
                                                 f16* __restrict__ S, int tid) {
  #pragma unroll
  for (int cc = 0; cc < 2; ++cc) {
    int ch = tid + cc * 256;            // 512 chunks of 8 f16 (16B)
    int r  = ch >> 2;
    int kq = (ch & 3) << 3;
    async_copy16(G + (row0 + r) * ld + k0 + kq, S + r * TKP + kq);
  }
}

// Guarded path for edge tiles (partial rows / K tail): via VGPRs + ds_store.
__device__ __forceinline__ void g2lds_tile(const f16* __restrict__ G, long ld,
                                           long row0, long maxRow, int k0, int K,
                                           f16* __restrict__ S, int tid) {
  #pragma unroll
  for (int cc = 0; cc < 2; ++cc) {
    int ch = tid + cc * 256;
    int r  = ch >> 2;
    int kq = (ch & 3) << 3;
    uint4 val = make_uint4(0u, 0u, 0u, 0u);
    long gr = row0 + r;
    int  gk = k0 + kq;
    if (gr < maxRow) {
      if (gk + 8 <= K) {
        val = *(const uint4*)(G + gr * ld + gk);
      } else {
        alignas(16) f16 tmp[8] = {};
        for (int e = 0; e < 8; ++e) if (gk + e < K) tmp[e] = G[gr * ld + gk + e];
        val = *(const uint4*)tmp;
      }
    }
    *(uint4*)(S + r * TKP + kq) = val;
  }
}

// ---------------------------------------------------------------------------
// WMMA GEMM: C[M,N] = epi( A[M,K](f16, lda) * B[N,K](f16 pre-transposed, ldb) )
// Block 128x128, 8 waves (2x4), each wave 64x32 = 4x2 WMMA 16x16 tiles.
// Tiles staged via async global->LDS (ASYNCcnt) on the full-tile fast path.
// Epilogue: +bias[n]; *bnScale[n]+bnShift[n]; act; +resid[m,n]; f32 or f16 out.
// ---------------------------------------------------------------------------
__global__ __launch_bounds__(256) void k_gemm(
    const f16* __restrict__ A, int lda,
    const f16* __restrict__ B, int ldb,
    void* __restrict__ Cv, int ldc,
    long M, int N, int K,
    const float* __restrict__ bias,
    const float* __restrict__ bnG, const float* __restrict__ bnB,
    const float* __restrict__ resid, int ldr,
    int act, int f16out)
{
  __shared__ f16 As[2][128 * TKP];
  __shared__ f16 Bs[2][128 * TKP];

  const int tid  = threadIdx.x;
  const int wave = tid >> 5, lane = tid & 31;
  const int wm = wave >> 2, wn = wave & 3;       // 2 x 4 wave grid
  const int lr = lane & 15, hi = lane >> 4;

  const long rowA0 = (long)blockIdx.y * 128;
  const long rowB0 = (long)blockIdx.x * 128;
  const bool fullA = (rowA0 + 128 <= M);
  const bool fullB = (rowB0 + 128 <= (long)N);

  v8f acc[4][2];
  #pragma unroll
  for (int mt = 0; mt < 4; ++mt)
    #pragma unroll
    for (int nt = 0; nt < 2; ++nt) acc[mt][nt] = (v8f)0.0f;

  const int nk = (K + 31) >> 5;
  {
    const bool fullK0 = (32 <= K);
    if (fullA && fullK0) g2lds_tile_async(A, lda, rowA0, 0, As[0], tid);
    else                 g2lds_tile(A, lda, rowA0, M, 0, K, As[0], tid);
    if (fullB && fullK0) g2lds_tile_async(B, ldb, rowB0, 0, Bs[0], tid);
    else                 g2lds_tile(B, ldb, rowB0, N, 0, K, Bs[0], tid);
  }
  wait_async0();
  __syncthreads();

  for (int kt = 0; kt < nk; ++kt) {
    const int cur = kt & 1;
    if (kt + 2 < nk) {     // L2 prefetch of the tile after next (speculative)
      __builtin_prefetch(A + rowA0 * (long)lda + (kt + 2) * 32, 0, 1);
      __builtin_prefetch(B + rowB0 * (long)ldb + (kt + 2) * 32, 0, 1);
    }
    if (kt + 1 < nk) {
      const int k0 = (kt + 1) << 5;
      const bool fullK = (k0 + 32 <= K);
      if (fullA && fullK) g2lds_tile_async(A, lda, rowA0, k0, As[cur ^ 1], tid);
      else                g2lds_tile(A, lda, rowA0, M, k0, K, As[cur ^ 1], tid);
      if (fullB && fullK) g2lds_tile_async(B, ldb, rowB0, k0, Bs[cur ^ 1], tid);
      else                g2lds_tile(B, ldb, rowB0, N, k0, K, Bs[cur ^ 1], tid);
    }
    // Build fragments per CDNA5 16-bit WMMA VGPR layouts.
    union Frag { half16 v; uint4 q[2]; };
    Frag fa[4], fb[2];
    const f16* Sa = As[cur];
    const f16* Sb = Bs[cur];
    const int aK0 = hi * 8;    // lanes 0-15: K0..7 & K16..23 ; 16-31: K8..15 & K24..31
    const int bK0 = hi * 8;    // lanes 0-15: K0..7 & K16..23 ; 16-31: K8..15 & K24..31
    #pragma unroll
    for (int mt = 0; mt < 4; ++mt) {
      const f16* p = Sa + (wm * 64 + mt * 16 + lr) * TKP;
      fa[mt].q[0] = *(const uint4*)(p + aK0);
      fa[mt].q[1] = *(const uint4*)(p + aK0 + 16);
    }
    #pragma unroll
    for (int nt = 0; nt < 2; ++nt) {
      const f16* p = Sb + (wn * 32 + nt * 16 + lr) * TKP;
      fb[nt].q[0] = *(const uint4*)(p + bK0);
      fb[nt].q[1] = *(const uint4*)(p + bK0 + 16);
    }
    #pragma unroll
    for (int mt = 0; mt < 4; ++mt)
      #pragma unroll
      for (int nt = 0; nt < 2; ++nt)
        acc[mt][nt] = __builtin_amdgcn_wmma_f32_16x16x32_f16(
            false, fa[mt].v, false, fb[nt].v, (short)0, acc[mt][nt], false, false);
    wait_async0();     // this wave's async fills of buf^1 are in LDS
    __syncthreads();   // all waves done reading cur & filling buf^1
  }

  // Epilogue. C/D layout: lane 0-15 -> N=lane, M=r; lane 16-31 -> N=lane-16, M=8+r.
  const long baseM = rowA0 + wm * 64;
  const int  baseN = (int)rowB0 + wn * 32;
  const float bnScaleK = rsqrtf(1.00001f);
  #pragma unroll
  for (int mt = 0; mt < 4; ++mt) {
    #pragma unroll
    for (int nt = 0; nt < 2; ++nt) {
      int n = baseN + nt * 16 + lr;
      if (n >= N) continue;
      float bv = bias ? bias[n] : 0.f;
      float sc = 1.f, sh = 0.f;
      if (bnG) { sc = bnG[n] * bnScaleK; sh = bnB[n]; }
      long m0 = baseM + mt * 16 + hi * 8;
      #pragma unroll
      for (int r = 0; r < 8; ++r) {
        long m = m0 + r;
        if (m >= M) continue;
        float v = acc[mt][nt][r] + bv;
        if (bnG) v = v * sc + sh;
        v = apply_act(v, act);
        if (resid) v += resid[m * (long)ldr + n];
        if (f16out) ((f16*)Cv)[m * (long)ldc + n] = (f16)v;
        else        ((float*)Cv)[m * (long)ldc + n] = v;
      }
    }
  }
}

// ---------------------------------------------------------------------------
// f32 (KxN) -> f16 (NxK) tiled transpose (for K-major weights -> N-major)
// ---------------------------------------------------------------------------
__global__ __launch_bounds__(256) void k_transpose(const float* __restrict__ src,
                                                   f16* __restrict__ dst, int K, int N) {
  __shared__ float tile[32][33];
  int kb = blockIdx.y * 32, nb = blockIdx.x * 32;
  int tx = threadIdx.x, ty = threadIdx.y;
  for (int i = ty; i < 32; i += 8) {
    int k = kb + i, n = nb + tx;
    tile[i][tx] = (k < K && n < N) ? src[(size_t)k * N + n] : 0.f;
  }
  __syncthreads();
  for (int i = ty; i < 32; i += 8) {
    int n = nb + i, k = kb + tx;
    if (n < N && k < K) dst[(size_t)n * K + k] = (f16)tile[tx][i];
  }
}

// f32 (strided rows) -> f16 compact convert
__global__ void k_cvt(const float* __restrict__ src, long stride, int cols,
                      f16* __restrict__ dst, size_t total) {
  size_t i = (size_t)blockIdx.x * blockDim.x + threadIdx.x;
  if (i >= total) return;
  int c = (int)(i % cols); size_t r = i / cols;
  dst[i] = (f16)src[r * stride + c];
}

// ---------------------------------------------------------------------------
// Stage kernels
// ---------------------------------------------------------------------------
__global__ void k_diff(const float* __restrict__ in, float* __restrict__ out, size_t total) {
  size_t i = (size_t)blockIdx.x * blockDim.x + threadIdx.x;
  if (i >= total) return;
  int c = (int)(i % 3); size_t r = i / 3;
  int hw = (int)(r % 1024); size_t bt = r / 1024;
  int t = (int)(bt % 160); size_t b = bt / 160;
  size_t base = ((b * 3 + c) * 160 + t) * 1024 + hw;
  float v = in[base];
  out[i] = (t > 0) ? (v - in[base - 1024]) : 0.f;
}

// 3x3 spatial conv (3 in-ch) + BN + ReLU, NDHWC out. 1 block per pixel.
__global__ __launch_bounds__(192) void k_stem_conv1(const float* __restrict__ X,
    const float* __restrict__ w, const float* __restrict__ bias,
    const float* __restrict__ g, const float* __restrict__ bb, float* __restrict__ out) {
  __shared__ float patch[27];
  size_t pix = blockIdx.x;
  int wc = (int)(pix & 31), h = (int)((pix >> 5) & 31);
  size_t bt = pix >> 10;
  int tid = threadIdx.x;
  if (tid < 27) {
    int i = tid / 9, r = tid % 9, ky = r / 3, kx = r % 3;
    int hh = h + ky - 1, ww = wc + kx - 1;
    float v = 0.f;
    if (hh >= 0 && hh < 32 && ww >= 0 && ww < 32)
      v = X[(bt * 1024 + (size_t)hh * 32 + ww) * 3 + i];
    patch[tid] = v;
  }
  __syncthreads();
  int o = tid;
  float s = bias[o];
  #pragma unroll
  for (int i = 0; i < 3; ++i)
    #pragma unroll
    for (int r = 0; r < 9; ++r)
      s += patch[i * 9 + r] * w[(o * 3 + i) * 9 + r];
  s = s * (g[o] * rsqrtf(1.00001f)) + bb[o];
  out[pix * 192 + o] = fmaxf(s, 0.f);
}

// 2x2 spatial maxpool, NDHWC; optional fused BN+ReLU before pooling.
__global__ void k_bnpool(const float* __restrict__ in, const float* __restrict__ g,
                         const float* __restrict__ bsh, float* __restrict__ out,
                         int C, int Ho, int Wo, int doBN, size_t total) {
  size_t i = (size_t)blockIdx.x * blockDim.x + threadIdx.x;
  if (i >= total) return;
  int c = (int)(i % C); size_t r = i / C;
  int wo = (int)(r % Wo); r /= Wo;
  int ho = (int)(r % Ho); size_t bt = r / Ho;
  int Hi = Ho * 2, Wi = Wo * 2;
  float sc = 1.f, sh = 0.f;
  if (doBN) { sc = g[c] * rsqrtf(1.00001f); sh = bsh[c]; }
  float m = -1e30f;
  #pragma unroll
  for (int dy = 0; dy < 2; ++dy)
    #pragma unroll
    for (int dx = 0; dx < 2; ++dx) {
      float v = in[((bt * Hi + (2 * ho + dy)) * (size_t)Wi + (2 * wo + dx)) * C + c];
      if (doBN) v = fmaxf(v * sc + sh, 0.f);
      m = fmaxf(m, v);
    }
  out[i] = m;
}

// Grouped depth-3 conv (INN expand): C in -> 2C out, +bias. NDHWC.
__global__ void k_expconv(const float* __restrict__ in, const float* __restrict__ w,
                          const float* __restrict__ bias, float* __restrict__ out,
                          int C, int T, int HW, size_t total) {
  size_t i = (size_t)blockIdx.x * blockDim.x + threadIdx.x;
  if (i >= total) return;
  int C2 = 2 * C;
  int oc = (int)(i % C2); size_t p = i / C2;
  int s = (int)(p % HW); size_t bt = p / HW;
  int t = (int)(bt % T); size_t b = bt / T;
  int cin = oc >> 1;
  float acc = bias[oc];
  #pragma unroll
  for (int d = 0; d < 3; ++d) {
    int tt = t + d - 1;
    if (tt >= 0 && tt < T)
      acc += in[(((b * T + tt) * (size_t)HW) + s) * C + cin] * w[oc * 3 + d];
  }
  out[i] = acc;
}

// Depthwise depth-3 conv + ReLU, f16 in/out.
__global__ void k_dw3(const f16* __restrict__ in, const float* __restrict__ w,
                      f16* __restrict__ out, int C, int T, int HW, size_t total) {
  size_t i = (size_t)blockIdx.x * blockDim.x + threadIdx.x;
  if (i >= total) return;
  int c = (int)(i % C); size_t p = i / C;
  int s = (int)(p % HW); size_t bt = p / HW;
  int t = (int)(bt % T); size_t b = bt / T;
  float acc = 0.f;
  #pragma unroll
  for (int d = 0; d < 3; ++d) {
    int tt = t + d - 1;
    if (tt >= 0 && tt < T)
      acc += (float)in[(((b * T + tt) * (size_t)HW) + s) * C + c] * w[c * 3 + d];
  }
  out[i] = (f16)fmaxf(acc, 0.f);
}

// x1 = x1 * exp(e2) + e3   (x1 is channel slice [.,0:C) of [pix][2C])
__global__ void k_inn_fuse(float* __restrict__ x, const float* __restrict__ e2,
                           const float* __restrict__ e3, int C, size_t total) {
  size_t i = (size_t)blockIdx.x * blockDim.x + threadIdx.x;
  if (i >= total) return;
  int c = (int)(i % C); size_t p = i / C;
  size_t idx = p * (size_t)(2 * C) + c;
  x[idx] = x[idx] * expf(e2[i]) + e3[i];
}

// Patch embed im2col: A[token][c*4+d] from P3 [b][160][4][4][768]
__global__ void k_patch_repack(const float* __restrict__ P3, f16* __restrict__ A, size_t total) {
  size_t i = (size_t)blockIdx.x * blockDim.x + threadIdx.x;
  if (i >= total) return;
  int k = (int)(i % 3072); size_t tok = i / 3072;
  int d = k & 3, c = k >> 2;
  int p = (int)(tok % 16); size_t bt = tok / 16;
  int t = (int)(bt % 40); size_t b = bt / 40;
  A[i] = (f16)P3[(((b * 160) + (size_t)(4 * t + d)) * 16 + p) * 768 + c];
}

__global__ void k_addpos(float* __restrict__ x, const float* __restrict__ pos,
                         const float* __restrict__ tpos, size_t total) {
  size_t i = (size_t)blockIdx.x * blockDim.x + threadIdx.x;
  if (i >= total) return;
  int c = (int)(i % 768); size_t tok = i / 768;
  int p = (int)(tok % 16); int t = (int)((tok / 16) % 40);
  x[i] += (pos[p * 768 + c] + tpos[t * 768 + c]) * (1.f / 255.f);
}

__device__ __forceinline__ float blockReduceSum(float v) {
  __shared__ float red[8];
  int lane = threadIdx.x & 31, w = threadIdx.x >> 5;
  #pragma unroll
  for (int o = 16; o > 0; o >>= 1) v += __shfl_down(v, o, 32);
  if (lane == 0) red[w] = v;
  __syncthreads();
  v = (threadIdx.x < 8) ? red[threadIdx.x] : 0.f;
  if (w == 0) {
    #pragma unroll
    for (int o = 4; o > 0; o >>= 1) v += __shfl_down(v, o, 32);
  }
  return v;
}

__global__ __launch_bounds__(256) void k_rmsnorm(const float* __restrict__ x,
    const float* __restrict__ w, f16* __restrict__ out, int C) {
  size_t row = blockIdx.x;
  const float* xr = x + row * C;
  float s = 0.f;
  for (int c = threadIdx.x; c < C; c += 256) { float v = xr[c]; s += v * v; }
  s = blockReduceSum(s);
  __shared__ float inv;
  if (threadIdx.x == 0) inv = rsqrtf(s / C + 1e-5f);
  __syncthreads();
  float iv = inv;
  for (int c = threadIdx.x; c < C; c += 256)
    out[row * C + c] = (f16)(xr[c] * w[c] * iv);
}

// Causal 4-tap depthwise conv over L + bias + SiLU. xz cols [0,1536).
__global__ void k_conv1d(const float* __restrict__ xz, const float* __restrict__ w,
                         const float* __restrict__ b, float* __restrict__ xc,
                         f16* __restrict__ xcb, size_t total) {
  size_t i = (size_t)blockIdx.x * blockDim.x + threadIdx.x;
  if (i >= total) return;
  int c = (int)(i % 1536); size_t tok = i / 1536;
  int l = (int)(tok % 640); size_t bb = tok / 640;
  float s = b[c];
  #pragma unroll
  for (int j = 0; j < 4; ++j) {
    int u = l - 3 + j;
    if (u >= 0) s += xz[((bb * 640 + u) * 3072) + c] * w[c * 4 + j];
  }
  float sv = s / (1.f + expf(-s));
  xc[i] = sv; xcb[i] = (f16)sv;
}

// Selective scan: one thread per (batch, channel). 16 states, L=640.
__global__ void k_scan(const float* __restrict__ xz, const float* __restrict__ xc,
                       const float* __restrict__ dt, const float* __restrict__ xdb,
                       const float* __restrict__ Alog, const float* __restrict__ Dp,
                       float* __restrict__ y, f16* __restrict__ yb, int L) {
  int idx = blockIdx.x * blockDim.x + threadIdx.x;
  if (idx >= 2 * 1536) return;
  int b = idx / 1536, c = idx % 1536;
  float Ac[16], h[16];
  #pragma unroll
  for (int n = 0; n < 16; ++n) { Ac[n] = -expf(Alog[c * 16 + n]); h[n] = 0.f; }
  float D = Dp[c];
  for (int t = 0; t < L; ++t) {
    size_t tok = (size_t)b * L + t;
    float dv = dt[tok * 1536 + c];
    float xv = xc[tok * 1536 + c];
    const float* BC = xdb + tok * 80;
    float yv = 0.f;
    #pragma unroll
    for (int n = 0; n < 16; ++n) {
      h[n] = expf(dv * Ac[n]) * h[n] + dv * BC[48 + n] * xv;
      yv += h[n] * BC[64 + n];
    }
    yv += xv * D;
    float z = xz[tok * 3072 + 1536 + c];
    yv *= z / (1.f + expf(-z));
    y[tok * 1536 + c] = yv;
    yb[tok * 1536 + c] = (f16)yv;
  }
}

// Signal-head im2col with x2 temporal repeat + zero pad.
__global__ void k_sig_repack(const float* __restrict__ x, f16* __restrict__ A,
                             int Tout, int Tin, size_t total) {
  size_t i = (size_t)blockIdx.x * blockDim.x + threadIdx.x;
  if (i >= total) return;
  int k = (int)(i % 2304); size_t tok = i / 2304;
  int d = k % 3, c = k / 3;
  int p = (int)(tok % 16); size_t bt = tok / 16;
  int t = (int)(bt % Tout); size_t b = bt / Tout;
  int u = t + d - 1;
  float v = 0.f;
  if (u >= 0 && u < Tout)
    v = x[(((b * Tin) + (size_t)(u >> 1)) * 16 + p) * 768 + c];
  A[i] = (f16)v;
}

__global__ __launch_bounds__(256) void k_final(const float* __restrict__ y2,
    const float* __restrict__ csw, const float* __restrict__ csb, float* __restrict__ out) {
  size_t bt = blockIdx.x;
  float s = 0.f;
  for (int i = threadIdx.x; i < 16 * 384; i += 256) {
    int p = i / 384, c = i % 384;
    s += y2[(bt * 16 + p) * 384 + c] * csw[c];
  }
  s = blockReduceSum(s);
  if (threadIdx.x == 0) out[bt] = s * (1.f / 16.f) + csb[0];
}

// ---------------------------------------------------------------------------
// Host-side orchestration
// ---------------------------------------------------------------------------
#define GRID1D(total) dim3((unsigned)(((total) + 255) / 256)), dim3(256), 0

static void gemmF(hipStream_t st, const f16* A, int lda, const f16* B, int ldb,
                  void* C, int ldc, long M, int N, int K,
                  const float* bias, const float* bnG, const float* bnB,
                  const float* res, int ldr, int act, int f16out) {
  dim3 g((unsigned)((N + 127) / 128), (unsigned)((M + 127) / 128)), b(256);
  k_gemm<<<g, b, 0, st>>>(A, lda, B, ldb, C, ldc, M, N, K, bias, bnG, bnB, res, ldr, act, f16out);
}
static void transposeW(hipStream_t st, const float* src, f16* dst, int K, int N) {
  k_transpose<<<dim3((unsigned)((N + 31) / 32), (unsigned)((K + 31) / 32)), dim3(32, 8), 0, st>>>(src, dst, K, N);
}
static void cvtW(hipStream_t st, const float* src, long stride, int cols, size_t rows, f16* dst) {
  size_t total = rows * (size_t)cols;
  k_cvt<<<GRID1D(total), st>>>(src, stride, cols, dst, total);
}

// One INN block: xe split x1/x2; x2 += dps1(x1); x1 = x1*exp(dps2(x2)) + dps3(x2)
static void run_inn(hipStream_t st, float* XE, int C, size_t pix, int T, int HW,
                    const float* const* P /* [dw,pw1,pw2]x3, exp_b, exp_w */,
                    f16* bfX, f16* tA, f16* tB, float* tC, float* tD, f16* WT) {
  size_t pc = pix * (size_t)C;
  // dps(x1) -> residual into x2
  cvtW(st, XE, 2 * C, C, pix, bfX);
  cvtW(st, P[1], C, C, C, WT);
  gemmF(st, bfX, C, WT, C, tA, C, (long)pix, C, C, nullptr, nullptr, nullptr, nullptr, 0, 1, 1);
  k_dw3<<<GRID1D(pc), st>>>(tA, P[0], tB, C, T, HW, pc);
  cvtW(st, P[2], C, C, C, WT);
  gemmF(st, tB, C, WT, C, XE + C, 2 * C, (long)pix, C, C, nullptr, nullptr, nullptr, XE + C, 2 * C, 0, 0);
  // dps2(x2) -> tC ; dps3(x2) -> tD
  cvtW(st, XE + C, 2 * C, C, pix, bfX);
  cvtW(st, P[4], C, C, C, WT);
  gemmF(st, bfX, C, WT, C, tA, C, (long)pix, C, C, nullptr, nullptr, nullptr, nullptr, 0, 1, 1);
  k_dw3<<<GRID1D(pc), st>>>(tA, P[3], tB, C, T, HW, pc);
  cvtW(st, P[5], C, C, C, WT);
  gemmF(st, tB, C, WT, C, tC, C, (long)pix, C, C, nullptr, nullptr, nullptr, nullptr, 0, 0, 0);
  cvtW(st, P[7], C, C, C, WT);
  gemmF(st, bfX, C, WT, C, tA, C, (long)pix, C, C, nullptr, nullptr, nullptr, nullptr, 0, 1, 1);
  k_dw3<<<GRID1D(pc), st>>>(tA, P[6], tB, C, T, HW, pc);
  cvtW(st, P[8], C, C, C, WT);
  gemmF(st, tB, C, WT, C, tD, C, (long)pix, C, C, nullptr, nullptr, nullptr, nullptr, 0, 0, 0);
  k_inn_fuse<<<GRID1D(pc), st>>>(XE, tC, tD, C, pc);
}

extern "C" void kernel_launch(void* const* d_in, const int* in_sizes, int n_in,
                              void* d_out, int out_size, void* d_ws, size_t ws_size,
                              hipStream_t stream) {
  (void)in_sizes; (void)n_in; (void)out_size; (void)ws_size;
  auto F = [&](int i) { return (const float*)d_in[i]; };
  // jax tree-flatten order (dict keys sorted): epoch, input_clip, params{...}
  const float* clip   = F(1);
  const float* mAlog  = F(2);  const float* mDp   = F(3);
  const float* mW1    = F(4);  const float* mW2   = F(5);
  const float* mWdt   = F(6);  const float* mWin  = F(7);
  const float* mWout  = F(8);  const float* mWx   = F(9);
  const float* mb1    = F(10); const float* mb2   = F(11); const float* mbdt  = F(12);
  const float* mconvb = F(13); const float* mconvw = F(14);
  const float* mn2w   = F(15); const float* mnw   = F(16);
  const float* patchb = F(17); const float* patchw = F(18); const float* pos = F(19);
  const float* sbn1b = F(20); const float* sbn1g = F(21);
  const float* sbn2b = F(22); const float* sbn2g = F(23);
  const float* sc1b  = F(24); const float* sc1w  = F(25);
  const float* sc2b  = F(26); const float* sc2w  = F(27);
  const float* cs_b  = F(28); const float* cs_w  = F(29);
  const float* tbn1b = F(30); const float* tbn1g = F(31);
  const float* tbn2b = F(32); const float* tbn2g = F(33);
  const float* tbn3b = F(34); const float* tbn3g = F(35);
  const float* tc1b  = F(36); const float* tc1w  = F(37);
  const float* i1p[11]; for (int j = 0; j < 11; ++j) i1p[j] = F(38 + j); // I1.dw,pw1,pw2, I2..., I3..., exp_b, exp_w
  const float* i2p[11]; for (int j = 0; j < 11; ++j) i2p[j] = F(49 + j);
  const float* tpos = F(60);

  size_t off = 0;
  auto WS = [&](size_t bytes) { void* p = (char*)d_ws + off; off += (bytes + 255) & ~(size_t)255; return p; };

  const size_t PIX1 = 2ull * 160 * 256;   // 16x16
  const size_t PIX2 = 2ull * 160 * 64;    // 8x8
  const size_t PIX3 = 2ull * 160 * 16;    // 4x4
  float* X0  = (float*)WS(2ull * 160 * 1024 * 3 * 4);
  float* S1  = (float*)WS(2ull * 160 * 1024 * 192 * 4);
  float* P1  = (float*)WS(PIX1 * 192 * 4);
  float* XE1 = (float*)WS(PIX1 * 384 * 4);
  f16*   bfX = (f16*)WS(PIX1 * 192 * 2);
  f16*   tA  = (f16*)WS(PIX1 * 192 * 2);
  f16*   tB  = (f16*)WS(PIX1 * 192 * 2);
  float* tC  = (float*)WS(PIX1 * 192 * 4);
  float* tD  = (float*)WS(PIX1 * 192 * 4);
  float* P2  = (float*)WS(PIX2 * 384 * 4);
  float* XE2 = (float*)WS(PIX2 * 768 * 4);
  float* P3  = (float*)WS(PIX3 * 768 * 4);
  f16*   Ap  = (f16*)WS(1280ull * 3072 * 2);
  f16*   WT  = (f16*)WS(3072ull * 768 * 2);
  float* xb  = (float*)WS(1280ull * 768 * 4);   // token stream
  f16*   hN  = (f16*)WS(1280ull * 768 * 2);
  float* xz  = (float*)WS(1280ull * 3072 * 4);
  float* xc  = (float*)WS(1280ull * 1536 * 4);
  f16*   xcb = (f16*)WS(1280ull * 1536 * 2);
  float* xdb = (float*)WS(1280ull * 80 * 4);
  f16*   xdbb = (f16*)WS(1280ull * 80 * 2);
  float* dtb = (float*)WS(1280ull * 1536 * 4);
  float* yv  = (float*)WS(1280ull * 1536 * 4);
  f16*   yb  = (f16*)WS(1280ull * 1536 * 2);
  f16*   ffb = (f16*)WS(1280ull * 3072 * 2);
  f16*   A1  = (f16*)WS(2560ull * 2304 * 2);
  float* y1  = (float*)WS(2560ull * 768 * 4);
  f16*   A2  = (f16*)WS(5120ull * 2304 * 2);
  float* y2  = (float*)WS(5120ull * 384 * 4);

  // ---- Stem ----
  {
    size_t t = 2ull * 160 * 1024 * 3;
    k_diff<<<GRID1D(t), stream>>>(clip, X0, t);
  }
  k_stem_conv1<<<dim3((unsigned)(2ull * 160 * 1024)), dim3(192), 0, stream>>>(X0, tc1w, tc1b, tbn1g, tbn1b, S1);
  { size_t t = PIX1 * 192;
    k_bnpool<<<GRID1D(t), stream>>>(S1, nullptr, nullptr, P1, 192, 16, 16, 0, t); }
  { size_t t = PIX1 * 384;
    k_expconv<<<GRID1D(t), stream>>>(P1, i1p[10], i1p[9], XE1, 192, 160, 256, t); }
  run_inn(stream, XE1, 192, PIX1, 160, 256, i1p, bfX, tA, tB, tC, tD, WT);
  { size_t t = PIX2 * 384;
    k_bnpool<<<GRID1D(t), stream>>>(XE1, tbn2g, tbn2b, P2, 384, 8, 8, 1, t); }
  { size_t t = PIX2 * 768;
    k_expconv<<<GRID1D(t), stream>>>(P2, i2p[10], i2p[9], XE2, 384, 160, 64, t); }
  run_inn(stream, XE2, 384, PIX2, 160, 64, i2p, bfX, tA, tB, tC, tD, WT);
  { size_t t = PIX3 * 768;
    k_bnpool<<<GRID1D(t), stream>>>(XE2, tbn3g, tbn3b, P3, 768, 4, 4, 1, t); }

  // ---- Patch embed + positions ----
  { size_t t = 1280ull * 3072;
    k_patch_repack<<<GRID1D(t), stream>>>(P3, Ap, t); }
  cvtW(stream, patchw, 3072, 3072, 768, WT);  // (768,768,4) already N x K
  gemmF(stream, Ap, 3072, WT, 3072, xb, 768, 1280, 768, 3072, patchb, nullptr, nullptr, nullptr, 0, 0, 0);
  { size_t t = 1280ull * 768;
    k_addpos<<<GRID1D(t), stream>>>(xb, pos, tpos, t); }

  // ---- Mamba x12 ----
  for (int l = 0; l < 12; ++l) {
    const float* Alog = mAlog + (size_t)l * 1536 * 16;
    k_rmsnorm<<<dim3(1280), dim3(256), 0, stream>>>(xb, mnw + (size_t)l * 768, hN, 768);
    transposeW(stream, mWin + (size_t)l * 768 * 3072, WT, 768, 3072);
    gemmF(stream, hN, 768, WT, 768, xz, 3072, 1280, 3072, 768, nullptr, nullptr, nullptr, nullptr, 0, 0, 0);
    { size_t t = 1280ull * 1536;
      k_conv1d<<<GRID1D(t), stream>>>(xz, mconvw + (size_t)l * 1536 * 4, mconvb + (size_t)l * 1536, xc, xcb, t); }
    transposeW(stream, mWx + (size_t)l * 1536 * 80, WT, 1536, 80);
    gemmF(stream, xcb, 1536, WT, 1536, xdb, 80, 1280, 80, 1536, nullptr, nullptr, nullptr, nullptr, 0, 0, 0);
    cvtW(stream, xdb, 80, 80, 1280, xdbb);
    transposeW(stream, mWdt + (size_t)l * 48 * 1536, WT, 48, 1536);
    gemmF(stream, xdbb, 80, WT, 48, dtb, 1536, 1280, 1536, 48, mbdt + (size_t)l * 1536, nullptr, nullptr, nullptr, 0, 3, 0);
    k_scan<<<dim3(12), dim3(256), 0, stream>>>(xz, xc, dtb, xdb, Alog, mDp + (size_t)l * 1536, yv, yb, 640);
    transposeW(stream, mWout + (size_t)l * 1536 * 768, WT, 1536, 768);
    gemmF(stream, yb, 1536, WT, 1536, xb, 768, 1280, 768, 1536, nullptr, nullptr, nullptr, xb, 768, 0, 0);
    k_rmsnorm<<<dim3(1280), dim3(256), 0, stream>>>(xb, mn2w + (size_t)l * 768, hN, 768);
    transposeW(stream, mW1 + (size_t)l * 768 * 3072, WT, 768, 3072);
    gemmF(stream, hN, 768, WT, 768, ffb, 3072, 1280, 3072, 768, mb1 + (size_t)l * 3072, nullptr, nullptr, nullptr, 0, 2, 1);
    transposeW(stream, mW2 + (size_t)l * 3072 * 768, WT, 3072, 768);
    gemmF(stream, ffb, 3072, WT, 3072, xb, 768, 1280, 768, 3072, mb2 + (size_t)l * 768, nullptr, nullptr, xb, 768, 0, 0);
  }

  // ---- Signal head ----
  { size_t t = 2560ull * 2304;
    k_sig_repack<<<GRID1D(t), stream>>>(xb, A1, 80, 40, t); }
  cvtW(stream, sc1w, 2304, 2304, 768, WT);
  gemmF(stream, A1, 2304, WT, 2304, y1, 768, 2560, 768, 2304, sc1b, sbn1g, sbn1b, nullptr, 0, 5, 0);
  { size_t t = 5120ull * 2304;
    k_sig_repack<<<GRID1D(t), stream>>>(y1, A2, 160, 80, t); }
  cvtW(stream, sc2w, 2304, 2304, 384, WT);
  gemmF(stream, A2, 2304, WT, 2304, y2, 384, 5120, 384, 2304, sc2b, sbn2g, sbn2b, nullptr, 0, 5, 0);
  k_final<<<dim3(320), dim3(256), 0, stream>>>(y2, cs_w, cs_b, (float*)d_out);
}
